// PoetryLSTM_54357106098327
// MI455X (gfx1250) — matrix operand order repaired and checked by reading, not compiled, not run on hardware
//
#include <hip/hip_runtime.h>

// Problem dims (hardcoded from reference): B,T,V,E,H,L = 128,256,8000,128,256,2
#define BB 128
#define TT 256
#define VV 8000
#define EE 128
#define HH 256
#define G4 (4 * HH)            // 1024 gate columns
#define MROWS (TT * BB)        // 32768 rows for batched GEMMs

typedef __attribute__((ext_vector_type(16))) __bf16 v16bf;
typedef __attribute__((ext_vector_type(8)))  __bf16 v8bf;
typedef __attribute__((ext_vector_type(8)))  float  v8f;

// ---------------------------------------------------------------------------
// WMMA fragment loaders (wave32, 16x16x32 bf16)
// A fragment: 16(M) x 32(K), A row-major [*, lda].
//   lane L: row = m0 + L%16 ; holds K = {kb..kb+7} U {kb+16..kb+23}, kb=(L/16)*8
// B fragment: 32(K) x 16(N) taken from W[N,K] row-major (i.e. computes A @ W^T).
//   lane L: col row-of-W = n0 + L%16 ; holds K = {kb..kb+7} U {kb+16..kb+23}, kb=(L/16)*8
// ---------------------------------------------------------------------------
__device__ __forceinline__ v16bf load_a_frag(const __bf16* __restrict__ A,
                                             int lda, int m0, int kk, int lane) {
  const int row = m0 + (lane & 15);
  const int kb  = kk + ((lane >> 4) << 3);
  const __bf16* p = A + (size_t)row * lda + kb;
  v8bf lo = *(const v8bf*)(p);
  v8bf hi = *(const v8bf*)(p + 16);
  v16bf a;
#pragma unroll
  for (int i = 0; i < 8; ++i) { a[i] = lo[i]; a[i + 8] = hi[i]; }
  return a;
}

__device__ __forceinline__ v16bf load_b_frag(const __bf16* __restrict__ W,
                                             int ldw, int wrow0, int kk, int lane) {
  const int row = wrow0 + (lane & 15);
  const __bf16* p = W + (size_t)row * ldw + kk + ((lane >> 4) << 3);
  v8bf lo = *(const v8bf*)(p);
  v8bf hi = *(const v8bf*)(p + 16);
  v16bf b;
#pragma unroll
  for (int i = 0; i < 8; ++i) { b[i] = lo[i]; b[i + 8] = hi[i]; }
  return b;
}

__device__ __forceinline__ v8f wmma_bf16(v16bf a, v16bf b, v8f c) {
  return __builtin_amdgcn_wmma_f32_16x16x32_bf16(false, a, false, b,
                                                 (short)0, c, false, false);
}

__device__ __forceinline__ float sigf(float x) {
  return 1.0f / (1.0f + __expf(-x));
}

// ---------------------------------------------------------------------------
// GEMM:  C[M,N](f32) = A[M,K](bf16) @ W[N,K](bf16)^T + bias[N]
// K is compile-time -> chunk loop fully unrolls (no register-rotation moves,
// scheduler hoists loads, waits stay partial).
// Block = 256 threads = 8 waves stacked in M; wave tile 32M x 64N
// (2 A frags x 4 B frags = 8 WMMAs per 32-K chunk).
// SWAP=1: output row r (= t*BB + b) remapped to [B,T,V] layout.
// Requires M%256==0, N%64==0, K%32==0.
// ---------------------------------------------------------------------------
template <int K, int SWAP>
__global__ __launch_bounds__(256)
void gemm_bf16_wmma(const __bf16* __restrict__ A, const __bf16* __restrict__ W,
                    const float* __restrict__ bias, float* __restrict__ C,
                    int N) {
  const int lane = threadIdx.x & 31;
  const int wave = threadIdx.x >> 5;
  const int m0 = blockIdx.x * 256 + wave * 32;
  const int n0 = blockIdx.y * 64;

  v8f acc[2][4] = {};

  v16bf a0 = load_a_frag(A, K, m0,      0, lane);
  v16bf a1 = load_a_frag(A, K, m0 + 16, 0, lane);
  v16bf bfr[4];
#pragma unroll
  for (int j = 0; j < 4; ++j) bfr[j] = load_b_frag(W, K, n0 + 16 * j, 0, lane);

#pragma unroll
  for (int kk = 32; kk < K; kk += 32) {
    // Issue next chunk's loads before consuming current fragments; with full
    // unroll SSA renaming makes the "rotation" free.
    v16bf a0n = load_a_frag(A, K, m0,      kk, lane);
    v16bf a1n = load_a_frag(A, K, m0 + 16, kk, lane);
    v16bf bn[4];
#pragma unroll
    for (int j = 0; j < 4; ++j) bn[j] = load_b_frag(W, K, n0 + 16 * j, kk, lane);

#pragma unroll
    for (int j = 0; j < 4; ++j) {
      acc[0][j] = wmma_bf16(a0, bfr[j], acc[0][j]);
      acc[1][j] = wmma_bf16(a1, bfr[j], acc[1][j]);
    }
    a0 = a0n; a1 = a1n;
#pragma unroll
    for (int j = 0; j < 4; ++j) bfr[j] = bn[j];
  }
#pragma unroll
  for (int j = 0; j < 4; ++j) {
    acc[0][j] = wmma_bf16(a0, bfr[j], acc[0][j]);
    acc[1][j] = wmma_bf16(a1, bfr[j], acc[1][j]);
  }

  const int colh = lane & 15;
  const int hi   = lane >> 4;
#pragma unroll
  for (int s = 0; s < 2; ++s) {
#pragma unroll
    for (int j = 0; j < 4; ++j) {
      const int col = n0 + 16 * j + colh;
      const float bv = bias[col];
#pragma unroll
      for (int r = 0; r < 8; ++r) {
        const int row = m0 + 16 * s + r + 8 * hi;
        size_t off;
        if (SWAP) {
          const int bidx = row & (BB - 1);   // row = t*BB + b
          const int tidx = row >> 7;
          off = (size_t)bidx * TT * VV + (size_t)tidx * VV + col;
        } else {
          off = (size_t)row * N + col;
        }
        C[off] = acc[s][j][r] + bv;
      }
    }
  }
}

// ---------------------------------------------------------------------------
// Fused LSTM timestep: gates = xg[t] + h_prev @ w_hh^T, then cell update.
// Grid = HH/16 = 16 blocks of 128 threads (4 waves cover all B=128 rows,
// 32 rows per wave). Each wave: 8 accumulators = 2 row sub-tiles x 4 gates,
// fully-unrolled pipelined K loop, 64 WMMAs per wave over K=256.
// ---------------------------------------------------------------------------
__global__ __launch_bounds__(128)
void lstm_step_wmma(const __bf16* __restrict__ hprev,   // [B,H] bf16
                    const __bf16* __restrict__ w_hh,    // [4H,H] bf16
                    const float* __restrict__ xg_t,     // [B,4H] f32 (bias folded)
                    const float* __restrict__ xg_next,  // [B,4H] of t+1 (or null)
                    float* __restrict__ c_state,        // [B,H] in/out
                    __bf16* __restrict__ h_out_bf16,    // hs[t] : [B,H]
                    float* __restrict__ h_out_f32) {    // running hT : [B,H]
  const int lane = threadIdx.x & 31;
  const int wave = threadIdx.x >> 5;  // 0..3
  const int m0 = wave * 32;           // batch rows
  const int n0 = blockIdx.x * 16;     // column within each gate block
  const int colh = lane & 15;
  const int hi   = lane >> 4;

  // Accumulators seeded from this timestep's precomputed input gates.
  v8f acc[2][4];
#pragma unroll
  for (int s = 0; s < 2; ++s) {
#pragma unroll
    for (int g = 0; g < 4; ++g) {
      const int col = g * HH + n0 + colh;
#pragma unroll
      for (int r = 0; r < 8; ++r) {
        const int row = m0 + 16 * s + r + 8 * hi;
        acc[s][g][r] = xg_t[(size_t)row * G4 + col];
      }
    }
  }

  // Warm L2/L1 for the next timestep's xg slice (global_prefetch path).
  if (xg_next) {
    __builtin_prefetch(xg_next + (size_t)(m0 + colh) * G4 + n0, 0, 1);
    __builtin_prefetch(xg_next + (size_t)(m0 + 16 + colh) * G4 + n0, 0, 1);
  }

  // K loop over H=256, compile-time bounds -> full unroll, no rotation moves.
  v16bf a0 = load_a_frag(hprev, HH, m0,      0, lane);
  v16bf a1 = load_a_frag(hprev, HH, m0 + 16, 0, lane);
  v16bf bfr[4];
#pragma unroll
  for (int g = 0; g < 4; ++g)
    bfr[g] = load_b_frag(w_hh, HH, g * HH + n0, 0, lane);

#pragma unroll
  for (int kk = 32; kk < HH; kk += 32) {
    v16bf a0n = load_a_frag(hprev, HH, m0,      kk, lane);
    v16bf a1n = load_a_frag(hprev, HH, m0 + 16, kk, lane);
    v16bf bn[4];
#pragma unroll
    for (int g = 0; g < 4; ++g)
      bn[g] = load_b_frag(w_hh, HH, g * HH + n0, kk, lane);

#pragma unroll
    for (int g = 0; g < 4; ++g) {
      acc[0][g] = wmma_bf16(a0, bfr[g], acc[0][g]);
      acc[1][g] = wmma_bf16(a1, bfr[g], acc[1][g]);
    }
    a0 = a0n; a1 = a1n;
#pragma unroll
    for (int g = 0; g < 4; ++g) bfr[g] = bn[g];
  }
#pragma unroll
  for (int g = 0; g < 4; ++g) {
    acc[0][g] = wmma_bf16(a0, bfr[g], acc[0][g]);
    acc[1][g] = wmma_bf16(a1, bfr[g], acc[1][g]);
  }

  // LSTM cell update, in-register per output element.
#pragma unroll
  for (int s = 0; s < 2; ++s) {
#pragma unroll
    for (int r = 0; r < 8; ++r) {
      const int row = m0 + 16 * s + r + 8 * hi;
      const size_t idx = (size_t)row * HH + n0 + colh;
      const float ig = sigf(acc[s][0][r]);
      const float fg = sigf(acc[s][1][r]);
      const float gg = tanhf(acc[s][2][r]);
      const float og = sigf(acc[s][3][r]);
      const float cn = fg * c_state[idx] + ig * gg;
      const float hn = og * tanhf(cn);
      c_state[idx]    = cn;
      h_out_bf16[idx] = (__bf16)hn;
      h_out_f32[idx]  = hn;
    }
  }
}

// ---------------------------------------------------------------------------
// Prep / glue kernels
// ---------------------------------------------------------------------------
__global__ void cast_f32_to_bf16(const float* __restrict__ in,
                                 __bf16* __restrict__ out, int n) {
  int i = blockIdx.x * 256 + threadIdx.x;
  if (i < n) out[i] = (__bf16)in[i];
}

__global__ void add_bias_vec(const float* __restrict__ a,
                             const float* __restrict__ b,
                             float* __restrict__ out, int n) {
  int i = blockIdx.x * 256 + threadIdx.x;
  if (i < n) out[i] = a[i] + b[i];
}

// x[B,T] (int indices) -> x_bf16[T,B,E] time-major gather from emb_bf16[V,E]
__global__ void embed_gather(const int* __restrict__ x,
                             const __bf16* __restrict__ embb,
                             __bf16* __restrict__ xb) {
  int tid = blockIdx.x * 256 + threadIdx.x;
  if (tid >= TT * BB * EE) return;
  const int t = tid / (BB * EE);
  const int rem = tid - t * (BB * EE);
  const int b = rem / EE;
  const int e = rem - b * EE;
  const int idx = x[b * TT + t];
  xb[tid] = embb[(size_t)idx * EE + e];
}

// h0[L,B,H] -> bf16 per-layer buffers; c0[L,B,H] -> f32 running c state
__global__ void init_state(const float* __restrict__ h0,
                           const float* __restrict__ c0,
                           __bf16* __restrict__ h0b,
                           float* __restrict__ cbuf) {
  int i = blockIdx.x * 256 + threadIdx.x;
  if (i < 2 * BB * HH) {
    h0b[i]  = (__bf16)h0[i];
    cbuf[i] = c0[i];
  }
}

// Append hN and cN after logits in d_out
__global__ void finalize_states(const float* __restrict__ hT,   // [2,B,H]
                                const float* __restrict__ cT,   // [2,B,H]
                                float* __restrict__ out) {
  int i = blockIdx.x * 256 + threadIdx.x;
  if (i < 2 * BB * HH) {
    const size_t base = (size_t)BB * TT * VV;
    out[base + i]               = hT[i];
    out[base + 2 * BB * HH + i] = cT[i];
  }
}

// ---------------------------------------------------------------------------
extern "C" void kernel_launch(void* const* d_in, const int* in_sizes, int n_in,
                              void* d_out, int out_size, void* d_ws, size_t ws_size,
                              hipStream_t stream) {
  (void)in_sizes; (void)n_in; (void)out_size; (void)ws_size;

  const int*   x      = (const int*)  d_in[0];
  const float* h0     = (const float*)d_in[1];
  const float* c0     = (const float*)d_in[2];
  const float* emb    = (const float*)d_in[3];
  const float* w_ih0  = (const float*)d_in[4];
  const float* w_hh0  = (const float*)d_in[5];
  const float* b_ih0  = (const float*)d_in[6];
  const float* b_hh0  = (const float*)d_in[7];
  const float* w_ih1  = (const float*)d_in[8];
  const float* w_hh1  = (const float*)d_in[9];
  const float* b_ih1  = (const float*)d_in[10];
  const float* b_hh1  = (const float*)d_in[11];
  const float* fc_w   = (const float*)d_in[12];
  const float* fc_b   = (const float*)d_in[13];
  float* out = (float*)d_out;

  // ---- carve workspace (256B aligned slabs) ----
  char* wp = (char*)d_ws;
  auto carve = [&](size_t bytes) -> void* {
    void* p = (void*)wp;
    wp += (bytes + 255) & ~(size_t)255;
    return p;
  };
  __bf16* embb   = (__bf16*)carve((size_t)VV * EE * 2);
  __bf16* wih0b  = (__bf16*)carve((size_t)G4 * EE * 2);
  __bf16* whh0b  = (__bf16*)carve((size_t)G4 * HH * 2);
  __bf16* wih1b  = (__bf16*)carve((size_t)G4 * HH * 2);
  __bf16* whh1b  = (__bf16*)carve((size_t)G4 * HH * 2);
  __bf16* fcwb   = (__bf16*)carve((size_t)VV * HH * 2);
  float*  bias0  = (float*) carve((size_t)G4 * 4);
  float*  bias1  = (float*) carve((size_t)G4 * 4);
  __bf16* xb     = (__bf16*)carve((size_t)MROWS * EE * 2);     // [T,B,E]
  float*  xg     = (float*) carve((size_t)MROWS * G4 * 4);     // reused L0 then L1
  __bf16* hs0b   = (__bf16*)carve((size_t)MROWS * HH * 2);     // [T,B,H]
  __bf16* hs1b   = (__bf16*)carve((size_t)MROWS * HH * 2);     // [T,B,H]
  __bf16* h0b    = (__bf16*)carve((size_t)2 * BB * HH * 2);    // initial h per layer
  float*  cbuf   = (float*) carve((size_t)2 * BB * HH * 4);    // running c per layer
  float*  hTbuf  = (float*) carve((size_t)2 * BB * HH * 4);    // running h per layer

  auto nb = [](int n) { return (n + 255) / 256; };

  // ---- precision casts + bias folding ----
  cast_f32_to_bf16<<<nb(VV * EE), 256, 0, stream>>>(emb,   embb,  VV * EE);
  cast_f32_to_bf16<<<nb(G4 * EE), 256, 0, stream>>>(w_ih0, wih0b, G4 * EE);
  cast_f32_to_bf16<<<nb(G4 * HH), 256, 0, stream>>>(w_hh0, whh0b, G4 * HH);
  cast_f32_to_bf16<<<nb(G4 * HH), 256, 0, stream>>>(w_ih1, wih1b, G4 * HH);
  cast_f32_to_bf16<<<nb(G4 * HH), 256, 0, stream>>>(w_hh1, whh1b, G4 * HH);
  cast_f32_to_bf16<<<nb(VV * HH), 256, 0, stream>>>(fc_w,  fcwb,  VV * HH);
  add_bias_vec<<<nb(G4), 256, 0, stream>>>(b_ih0, b_hh0, bias0, G4);
  add_bias_vec<<<nb(G4), 256, 0, stream>>>(b_ih1, b_hh1, bias1, G4);
  init_state<<<nb(2 * BB * HH), 256, 0, stream>>>(h0, c0, h0b, cbuf);

  // ---- embedding gather (time-major) ----
  embed_gather<<<nb(TT * BB * EE), 256, 0, stream>>>(x, embb, xb);

  // ---- layer 0: batched input gates, then sequential scan ----
  gemm_bf16_wmma<EE, 0><<<dim3(MROWS / 256, G4 / 64), 256, 0, stream>>>(
      xb, wih0b, bias0, xg, G4);
  for (int t = 0; t < TT; ++t) {
    const __bf16* hprev = (t == 0) ? h0b : (hs0b + (size_t)(t - 1) * BB * HH);
    const float* xgn = (t + 1 < TT) ? (xg + (size_t)(t + 1) * BB * G4) : nullptr;
    lstm_step_wmma<<<HH / 16, 128, 0, stream>>>(
        hprev, whh0b, xg + (size_t)t * BB * G4, xgn, cbuf,
        hs0b + (size_t)t * BB * HH, hTbuf);
  }

  // ---- layer 1 ----
  gemm_bf16_wmma<HH, 0><<<dim3(MROWS / 256, G4 / 64), 256, 0, stream>>>(
      hs0b, wih1b, bias1, xg, G4);
  for (int t = 0; t < TT; ++t) {
    const __bf16* hprev = (t == 0) ? (h0b + BB * HH)
                                   : (hs1b + (size_t)(t - 1) * BB * HH);
    const float* xgn = (t + 1 < TT) ? (xg + (size_t)(t + 1) * BB * G4) : nullptr;
    lstm_step_wmma<<<HH / 16, 128, 0, stream>>>(
        hprev, whh1b, xg + (size_t)t * BB * G4, xgn, cbuf + BB * HH,
        hs1b + (size_t)t * BB * HH, hTbuf + BB * HH);
  }

  // ---- output projection straight into d_out ([B,T,V] remap in stores) ----
  gemm_bf16_wmma<HH, 1><<<dim3(MROWS / 256, VV / 64), 256, 0, stream>>>(
      hs1b, fcwb, fc_b, out, VV);

  // ---- hN, cN appended after logits ----
  finalize_states<<<nb(2 * BB * HH), 256, 0, stream>>>(hTbuf, cbuf, out);
}
